// GCNModel_13692355740143
// MI455X (gfx1250) — compile-verified, matrix operand order checked by reading, not run on hardware
//
#include <hip/hip_runtime.h>

typedef __attribute__((ext_vector_type(16))) _Float16 v16h;
typedef __attribute__((ext_vector_type(8)))  _Float16 v8h;
typedef __attribute__((ext_vector_type(8)))  float    v8f;
typedef __attribute__((ext_vector_type(4)))  unsigned int u32x4;
typedef __attribute__((ext_vector_type(8)))  int      i32x8;
typedef __attribute__((ext_vector_type(4)))  int      i32x4;

#define IN_DIM 128
#define LDP    (IN_DIM + 8)   // padded LDS row stride in f16 (256B row + 16B pad)

// ---------------------------------------------------------------- utilities
__global__ void zero_f32(float* __restrict__ p, long long n) {
  long long i = (long long)blockIdx.x * blockDim.x + threadIdx.x;
  if (i < n) p[i] = 0.0f;
}

__global__ void count_edges(const int* __restrict__ col, float* __restrict__ cnt, int E) {
  int e = blockIdx.x * blockDim.x + threadIdx.x;
  if (e < E)
    (void)__hip_atomic_fetch_add(&cnt[col[e]], 1.0f, __ATOMIC_RELAXED,
                                 __HIP_MEMORY_SCOPE_AGENT);
}

// W: [128, out_dim] fp32 (row-major)  ->  Wt: [out_dim, 128] f16 (transposed)
__global__ void convert_w(const float* __restrict__ W, _Float16* __restrict__ Wt, int out_dim) {
  int i = blockIdx.x * blockDim.x + threadIdx.x;  // i = n*128 + k
  if (i >= out_dim * IN_DIM) return;
  int n = i >> 7, k = i & 127;
  Wt[i] = (_Float16)W[k * out_dim + n];
}

// ---------------------------------------------------------------- WMMA GEMM
// H[row, ncol] = sum_k X[row,k] * W[k,ncol] + bias[ncol]
// One block = 16 rows of X; one wave (32 lanes) per 16-column tile.
// Weight tile DMA'd to LDS by the Tensor Data Mover (padded rows, no bank
// conflicts); activation tile staged fp32->f16 by the block in parallel.
template <int OUT>
__global__ void gemm_bias(const float* __restrict__ X, const _Float16* __restrict__ Wt,
                          const float* __restrict__ bias, float* __restrict__ H, int nrows) {
  __shared__ __align__(16) _Float16 As[16 * LDP];
  __shared__ __align__(16) _Float16 Bs[OUT * LDP];

  const int tid = threadIdx.x;
  const int rowBase = blockIdx.x * 16;

#if __has_builtin(__builtin_amdgcn_tensor_load_to_lds) && \
    __has_builtin(__builtin_amdgcn_s_wait_tensorcnt)
  // --- TDM: async DMA Wt[OUT,128] f16 -> Bs with 16B pad per 256B row ---
  if (tid < 32) {
    unsigned long long ga = (unsigned long long)(uintptr_t)Wt;
    unsigned int la = (unsigned int)(uintptr_t)&Bs[0];  // flat low bits = LDS offset
    u32x4 g0;
    g0.x = 1u;                                              // count=1, user mode
    g0.y = la;                                              // lds_addr
    g0.z = (unsigned int)ga;                                // global_addr[31:0]
    g0.w = (unsigned int)((ga >> 32) & 0x01FFFFFFu) | 0x80000000u;  // [56:32] | type=2
    i32x8 g1;
    g1[0] = (int)((1u << 16)        // data_size = 2 bytes
                  | (1u << 20)      // pad_enable
                  | (5u << 22)      // pad_interval: 64 DWORDs (256B) per row
                  | (3u << 25));    // pad_amount: 4 DWORDs (16B)
    g1[1] = (int)(128u << 16);              // tensor_dim0 = 128 (bits 79:48)
    g1[2] = (int)((unsigned)OUT << 16);     // tensor_dim1 = OUT (bits 111:80)
    g1[3] = (int)(128u << 16);              // tile_dim0 = 128 (bits 127:112)
    g1[4] = (int)OUT;                       // tile_dim1 = OUT (bits 143:128)
    g1[5] = 128;                            // tensor_dim0_stride = 128
    g1[6] = 0;
    g1[7] = 0;
    i32x4 z4 = {0, 0, 0, 0};                // groups 2/3 unused (2-D tile)
    i32x8 z8 = {0, 0, 0, 0, 0, 0, 0, 0};
    __builtin_amdgcn_tensor_load_to_lds(g0, g1, z4, z4, z8, 0);
  }
#endif

  // Stage a 16x128 activation tile, fp32 -> f16 (overlaps with the TDM DMA)
  for (int i = tid; i < 16 * IN_DIM; i += blockDim.x) {
    int r = i >> 7, k = i & 127;
    int row = rowBase + r;
    float v = (row < nrows) ? X[(long long)row * IN_DIM + k] : 0.0f;
    As[r * LDP + k] = (_Float16)v;
  }

#if __has_builtin(__builtin_amdgcn_tensor_load_to_lds) && \
    __has_builtin(__builtin_amdgcn_s_wait_tensorcnt)
  if (tid < 32) __builtin_amdgcn_s_wait_tensorcnt(0);
#else
  for (int i = tid; i < OUT * IN_DIM; i += blockDim.x)
    Bs[(i >> 7) * LDP + (i & 127)] = Wt[i];
#endif
  __syncthreads();

  const int lane = tid & 31;
  const int wave = tid >> 5;   // column-tile index
  const int half = lane >> 4;  // 0: lanes 0-15 (K 0-7/16-23), 1: lanes 16-31 (K 8-15/24-31)
  const int ml   = lane & 15;

  const _Float16* arow = As + ml * LDP;                // row ml of A tile
  const _Float16* brow = Bs + (wave * 16 + ml) * LDP;  // column ncol of W

  v8f acc = {};
#pragma unroll
  for (int c = 0; c < IN_DIM / 32; ++c) {
    int kb = 32 * c + 8 * half;
    v8h alo = *(const v8h*)(arow + kb);        // K = kb .. kb+7   (VGPR 0-3)
    v8h ahi = *(const v8h*)(arow + kb + 16);   // K = kb+16..kb+23 (VGPR 4-7)
    v8h blo = *(const v8h*)(brow + kb);
    v8h bhi = *(const v8h*)(brow + kb + 16);
    v16h a = __builtin_shufflevector(alo, ahi, 0, 1, 2, 3, 4, 5, 6, 7,
                                     8, 9, 10, 11, 12, 13, 14, 15);
    v16h b = __builtin_shufflevector(blo, bhi, 0, 1, 2, 3, 4, 5, 6, 7,
                                     8, 9, 10, 11, 12, 13, 14, 15);
    acc = __builtin_amdgcn_wmma_f32_16x16x32_f16(false, a, false, b, (short)0, acc,
                                                 false, false);
  }

  // C/D layout: VGPR r -> row (r + 8*half), col = lane&15 within the tile
  const int ncol = wave * 16 + ml;
  const float bv = bias[ncol];
  float* hp = H + (long long)(rowBase + 8 * half) * OUT + ncol;
  if (rowBase + 16 <= nrows) {   // full tile: branch-free stores
#pragma unroll
    for (int r = 0; r < 8; ++r) hp[(long long)r * OUT] = acc[r] + bv;
  } else {
#pragma unroll
    for (int r = 0; r < 8; ++r) {
      int row = rowBase + r + 8 * half;
      if (row < nrows) H[(long long)row * OUT + ncol] = acc[r] + bv;
    }
  }
}

// ---------------------------------------------------------------- edge scatter
template <int DIM>
__global__ void scatter_edges(const float* __restrict__ H, const int* __restrict__ row,
                              const int* __restrict__ col, float* __restrict__ S, int E) {
  constexpr int LPE = DIM / 4;  // lanes per edge (float4 per lane)
  long long idx = (long long)blockIdx.x * blockDim.x + threadIdx.x;
  if (idx >= (long long)E * LPE) return;
  int e = (int)(idx / LPE);
  int c = ((int)(idx % LPE)) * 4;
  const float4 v = *(const float4*)(H + (long long)row[e] * DIM + c);
  float* dst = S + (long long)col[e] * DIM + c;
  (void)__hip_atomic_fetch_add(dst + 0, v.x, __ATOMIC_RELAXED, __HIP_MEMORY_SCOPE_AGENT);
  (void)__hip_atomic_fetch_add(dst + 1, v.y, __ATOMIC_RELAXED, __HIP_MEMORY_SCOPE_AGENT);
  (void)__hip_atomic_fetch_add(dst + 2, v.z, __ATOMIC_RELAXED, __HIP_MEMORY_SCOPE_AGENT);
  (void)__hip_atomic_fetch_add(dst + 3, v.w, __ATOMIC_RELAXED, __HIP_MEMORY_SCOPE_AGENT);
}

// ---------------------------------------------------------------- finalize
__global__ void finalize_mean_relu(const float* __restrict__ S, const float* __restrict__ H,
                                   const float* __restrict__ cnt, float* __restrict__ out,
                                   long long total, int dim) {
  long long i = (long long)blockIdx.x * blockDim.x + threadIdx.x;
  if (i >= total) return;
  long long r = i / dim;
  float c = cnt[r];
  float v = (c > 0.0f) ? S[i] / c : H[i];
  out[i] = fmaxf(v, 0.0f);
}

// mean/where + log_softmax over 64 cols; one wave32 per row, 2 cols per lane
__global__ void finalize_logsoftmax64(const float* __restrict__ S, const float* __restrict__ H,
                                      const float* __restrict__ cnt, float* __restrict__ out,
                                      int n) {
  int r = blockIdx.x * (blockDim.x >> 5) + (threadIdx.x >> 5);
  int lane = threadIdx.x & 31;
  if (r >= n) return;
  long long base = (long long)r * 64;
  float c = cnt[r];
  float v0, v1;
  if (c > 0.0f) {
    float inv = 1.0f / c;
    v0 = S[base + lane] * inv;
    v1 = S[base + lane + 32] * inv;
  } else {
    v0 = H[base + lane];
    v1 = H[base + lane + 32];
  }
  float m = fmaxf(v0, v1);
  for (int off = 16; off; off >>= 1) m = fmaxf(m, __shfl_xor(m, off, 32));
  float s = expf(v0 - m) + expf(v1 - m);
  for (int off = 16; off; off >>= 1) s += __shfl_xor(s, off, 32);
  float lse = m + logf(s);
  out[base + lane] = v0 - lse;
  out[base + lane + 32] = v1 - lse;
}

// ---------------------------------------------------------------- driver
extern "C" void kernel_launch(void* const* d_in, const int* in_sizes, int n_in,
                              void* d_out, int out_size, void* d_ws, size_t ws_size,
                              hipStream_t stream) {
  const float* x  = (const float*)d_in[0];
  const int*   ei = (const int*)d_in[1];
  const float* W1 = (const float*)d_in[2];
  const float* b1 = (const float*)d_in[3];
  const float* W2 = (const float*)d_in[4];
  const float* b2 = (const float*)d_in[5];
  const float* W3 = (const float*)d_in[6];
  const float* b3 = (const float*)d_in[7];

  const int N = in_sizes[0] / IN_DIM;
  const int E = in_sizes[1] / 2;
  const int* row = ei;
  const int* col = ei + E;

  // workspace layout
  char* ws = (char*)d_ws;
  float* A1  = (float*)ws;                       // [N,128] activations
  float* H   = A1 + (size_t)N * IN_DIM;          // [N,128] linear output
  float* S   = H + (size_t)N * IN_DIM;           // [N,128] scatter accumulator
  float* cnt = S + (size_t)N * IN_DIM;           // [N] in-degree
  size_t whOff = (((size_t)N * IN_DIM * 3 + N) * sizeof(float) + 255) & ~(size_t)255;
  _Float16* Wh = (_Float16*)(ws + whOff);        // [out,128] f16 transposed weights

  const long long nd128 = (long long)N * 128;
  const long long nd64  = (long long)N * 64;
  const int rowTiles = (N + 15) / 16;
  const int TB = 256;

  // in-degree counts (layer-invariant)
  zero_f32<<<(unsigned)((N + TB - 1) / TB), TB, 0, stream>>>(cnt, N);
  count_edges<<<(unsigned)((E + TB - 1) / TB), TB, 0, stream>>>(col, cnt, E);

  // ---------------- layer 1 (128 -> 128, relu)
  convert_w<<<(128 * 128 + TB - 1) / TB, TB, 0, stream>>>(W1, Wh, 128);
  gemm_bias<128><<<rowTiles, 256, 0, stream>>>(x, Wh, b1, H, N);
  zero_f32<<<(unsigned)((nd128 + TB - 1) / TB), TB, 0, stream>>>(S, nd128);
  {
    long long nwork = (long long)E * 32;
    scatter_edges<128><<<(unsigned)((nwork + TB - 1) / TB), TB, 0, stream>>>(H, row, col, S, E);
  }
  finalize_mean_relu<<<(unsigned)((nd128 + TB - 1) / TB), TB, 0, stream>>>(S, H, cnt, A1,
                                                                           nd128, 128);

  // ---------------- layer 2 (128 -> 128, relu)
  convert_w<<<(128 * 128 + TB - 1) / TB, TB, 0, stream>>>(W2, Wh, 128);
  gemm_bias<128><<<rowTiles, 256, 0, stream>>>(A1, Wh, b2, H, N);
  zero_f32<<<(unsigned)((nd128 + TB - 1) / TB), TB, 0, stream>>>(S, nd128);
  {
    long long nwork = (long long)E * 32;
    scatter_edges<128><<<(unsigned)((nwork + TB - 1) / TB), TB, 0, stream>>>(H, row, col, S, E);
  }
  finalize_mean_relu<<<(unsigned)((nd128 + TB - 1) / TB), TB, 0, stream>>>(S, H, cnt, A1,
                                                                           nd128, 128);

  // ---------------- layer 3 (128 -> 64, log_softmax)
  convert_w<<<(128 * 64 + TB - 1) / TB, TB, 0, stream>>>(W3, Wh, 64);
  gemm_bias<64><<<rowTiles, 128, 0, stream>>>(A1, Wh, b3, H, N);
  zero_f32<<<(unsigned)((nd64 + TB - 1) / TB), TB, 0, stream>>>(S, nd64);
  {
    long long nwork = (long long)E * 16;
    scatter_edges<64><<<(unsigned)((nwork + TB - 1) / TB), TB, 0, stream>>>(H, row, col, S, E);
  }
  finalize_logsoftmax64<<<(unsigned)((N + 7) / 8), 256, 0, stream>>>(S, H, cnt,
                                                                    (float*)d_out, N);
}
